// SEW_Block_Linear_9388798509211
// MI455X (gfx1250) — 16-bit and 8-bit matrix operand order repaired and checked by reading, compiled, not run on hardware
//
#include <hip/hip_runtime.h>
#include <hip/hip_bf16.h>


typedef __attribute__((ext_vector_type(8)))  _Float16 v8h;
typedef __attribute__((ext_vector_type(16))) _Float16 v16h;
typedef __attribute__((ext_vector_type(8)))  float    v8f;
typedef __attribute__((ext_vector_type(8)))  int      v8i;
typedef __attribute__((ext_vector_type(4)))  int      v4i;
typedef __attribute__((ext_vector_type(2)))  int      v2i;

#define TT 4
#define BB 32
#define NN 256
#define CC 512
#define HH 8
#define DD 64
#define HD 2048
#define MROWS (TT * BB * NN)
#define EE   ((size_t)BB * NN * CC)
#define EEH  ((size_t)BB * NN * HD)

__global__ void fold_bn_kernel(const float* __restrict__ W, const float* __restrict__ b,
                               const float* __restrict__ g, const float* __restrict__ be,
                               const float* __restrict__ mu, const float* __restrict__ var,
                               _Float16* __restrict__ Wt, float* __restrict__ bias,
                               int K, int N) {
  size_t idx = (size_t)blockIdx.x * 256 + threadIdx.x;
  if (idx >= (size_t)K * N) return;
  int n = (int)(idx / K);
  int k = (int)(idx % K);
  float s = g[n] * rsqrtf(var[n] + 1e-5f);
  Wt[(size_t)n * K + k] = (_Float16)(W[(size_t)k * N + n] * s);
  if (k == 0) bias[n] = (b[n] - mu[n]) * s + be[n];
}

__global__ void cvt_f16_kernel(const float* __restrict__ X, _Float16* __restrict__ Xh,
                               size_t n) {
  size_t i = (size_t)blockIdx.x * 256 + threadIdx.x;
  if (i < n) Xh[i] = (_Float16)X[i];
}

__global__ void __launch_bounds__(128)
gemm_f16_kernel(const _Float16* __restrict__ A, const _Float16* __restrict__ Wt,
                const float* __restrict__ bias, float* __restrict__ Y,
                int K, int Ncols) {
  const int lane = threadIdx.x;
  const int wave = threadIdx.y;
  const int r    = lane & 15;
  const int hf   = lane >> 4;
  const int tm = (blockIdx.x * 4 + wave) * 32;
  const int tn = blockIdx.y * 64;

  const _Float16* arow0 = A + (size_t)(tm + r) * K + hf * 16;
  const _Float16* arow1 = arow0 + (size_t)16 * K;

  v8f acc[2][4] = {};
  for (int k = 0; k < K; k += 32) {
    v8h a00 = *(const v8h*)(arow0 + k);
    v8h a01 = *(const v8h*)(arow0 + k + 8);
    v16h a0 = __builtin_shufflevector(a00, a01, 0,1,2,3,4,5,6,7,8,9,10,11,12,13,14,15);
    v8h a10 = *(const v8h*)(arow1 + k);
    v8h a11 = *(const v8h*)(arow1 + k + 8);
    v16h a1 = __builtin_shufflevector(a10, a11, 0,1,2,3,4,5,6,7,8,9,10,11,12,13,14,15);
    __builtin_prefetch(arow0 + k + 64, 0, 1);
    __builtin_prefetch(arow1 + k + 64, 0, 1);
#pragma unroll
    for (int j = 0; j < 4; ++j) {
      v16h b = *(const v16h*)(Wt + (size_t)(tn + j * 16 + r) * K + k + hf * 16);
      acc[0][j] = __builtin_amdgcn_wmma_f32_16x16x32_f16(
          false, a0, false, b, (short)0, acc[0][j], false, false);
      acc[1][j] = __builtin_amdgcn_wmma_f32_16x16x32_f16(
          false, a1, false, b, (short)0, acc[1][j], false, false);
    }
  }
#pragma unroll
  for (int j = 0; j < 4; ++j) {
    const int col = tn + j * 16 + r;
    const float bv = bias[col];
#pragma unroll
    for (int m = 0; m < 2; ++m) {
#pragma unroll
      for (int g = 0; g < 8; ++g) {
        const int row = tm + m * 16 + hf * 8 + g;
        Y[(size_t)row * Ncols + col] = acc[m][j][g] + bv;
      }
    }
  }
}

__global__ void lif_f16_kernel(const float* __restrict__ Y, _Float16* __restrict__ S,
                               float vth, size_t E) {
  size_t e = (size_t)blockIdx.x * 256 + threadIdx.x;
  if (e >= E) return;
  float v = 0.f;
#pragma unroll
  for (int t = 0; t < TT; ++t) {
    v += (Y[t * E + e] - v) * 0.5f;
    float s = (v >= vth) ? 1.f : 0.f;
    S[t * E + e] = (_Float16)s;
    v *= (1.f - s);
  }
}

__global__ void lif_u8_kernel(const float* __restrict__ Y, unsigned char* __restrict__ S,
                              float vth, size_t E) {
  size_t e = (size_t)blockIdx.x * 256 + threadIdx.x;
  if (e >= E) return;
  float v = 0.f;
#pragma unroll
  for (int t = 0; t < TT; ++t) {
    v += (Y[t * E + e] - v) * 0.5f;
    float s = (v >= vth) ? 1.f : 0.f;
    S[t * E + e] = (unsigned char)s;
    v *= (1.f - s);
  }
}

__global__ void lif_u8T_kernel(const float* __restrict__ Y, unsigned char* __restrict__ VT,
                               float vth) {
  size_t e = (size_t)blockIdx.x * 256 + threadIdx.x;
  if (e >= EE) return;
  int c = (int)(e % CC);
  int n = (int)((e / CC) % NN);
  int b = (int)(e / ((size_t)CC * NN));
  int h = c >> 6, d = c & 63;
  float v = 0.f;
#pragma unroll
  for (int t = 0; t < TT; ++t) {
    v += (Y[t * EE + e] - v) * 0.5f;
    float s = (v >= vth) ? 1.f : 0.f;
    VT[((((size_t)(t * BB + b)) * HH + h) * DD + d) * NN + n] = (unsigned char)s;
    v *= (1.f - s);
  }
}

__global__ void lif_add_kernel(const float* __restrict__ Y, const float* __restrict__ X,
                               float* __restrict__ X1, _Float16* __restrict__ X1h,
                               float vth, size_t E) {
  size_t e = (size_t)blockIdx.x * 256 + threadIdx.x;
  if (e >= E) return;
  float v = 0.f;
#pragma unroll
  for (int t = 0; t < TT; ++t) {
    v += (Y[t * E + e] - v) * 0.5f;
    float s = (v >= vth) ? 1.f : 0.f;
    float xv = X[t * E + e] + s;
    X1[t * E + e] = xv;
    X1h[t * E + e] = (_Float16)xv;
    v *= (1.f - s);
  }
}

__global__ void lif_final_kernel(const float* __restrict__ Y, const float* __restrict__ X1,
                                 float* __restrict__ Out, float vth, size_t E) {
  size_t e = (size_t)blockIdx.x * 256 + threadIdx.x;
  if (e >= E) return;
  float v = 0.f;
#pragma unroll
  for (int t = 0; t < TT; ++t) {
    v += (Y[t * E + e] - v) * 0.5f;
    float s = (v >= vth) ? 1.f : 0.f;
    Out[t * E + e] = X1[t * E + e] + s;
    v *= (1.f - s);
  }
}

__global__ void __launch_bounds__(256)
attn_iu8_kernel(const unsigned char* __restrict__ Q8, const unsigned char* __restrict__ K8,
                const unsigned char* __restrict__ VT8, float* __restrict__ O) {
  extern __shared__ unsigned char smem[];
  unsigned char* sQ = smem;
  unsigned char* sK = smem + 16384;
  unsigned char* sV = smem + 32768;
  unsigned char* sA = smem + 49152;

  const int tbh = blockIdx.x;
  const int h   = tbh & (HH - 1);
  const int tb  = tbh >> 3;
  const int tid = threadIdx.x;

  {
    const unsigned char* qg = Q8 + (size_t)tb * NN * CC + (size_t)tid * CC + h * DD;
    const unsigned char* kg = K8 + (size_t)tb * NN * CC + (size_t)tid * CC + h * DD;
    const unsigned char* vg = VT8 + (size_t)tbh * DD * NN + (size_t)tid * 64;
    unsigned ql = (unsigned)(uintptr_t)(sQ + tid * 64);
    unsigned kl = (unsigned)(uintptr_t)(sK + tid * 64);
    unsigned vl = (unsigned)(uintptr_t)(sV + tid * 64);
    asm volatile(
        "global_load_async_to_lds_b128 %0, %3, off\n\t"
        "global_load_async_to_lds_b128 %0, %3, off offset:16\n\t"
        "global_load_async_to_lds_b128 %0, %3, off offset:32\n\t"
        "global_load_async_to_lds_b128 %0, %3, off offset:48\n\t"
        "global_load_async_to_lds_b128 %1, %4, off\n\t"
        "global_load_async_to_lds_b128 %1, %4, off offset:16\n\t"
        "global_load_async_to_lds_b128 %1, %4, off offset:32\n\t"
        "global_load_async_to_lds_b128 %1, %4, off offset:48\n\t"
        "global_load_async_to_lds_b128 %2, %5, off\n\t"
        "global_load_async_to_lds_b128 %2, %5, off offset:16\n\t"
        "global_load_async_to_lds_b128 %2, %5, off offset:32\n\t"
        "global_load_async_to_lds_b128 %2, %5, off offset:48"
        :
        : "v"(ql), "v"(kl), "v"(vl), "v"(qg), "v"(kg), "v"(vg)
        : "memory");
    asm volatile("s_wait_asynccnt 0" ::: "memory");
  }
  __syncthreads();

  const int lane = tid & 31;
  const int wave = tid >> 5;
  const int r  = lane & 15;
  const int hf = lane >> 4;

  for (int tt = wave; tt < 256; tt += 8) {
    const int tmi = tt >> 4, tni = tt & 15;
    const unsigned char* qr = sQ + (size_t)(tmi * 16 + r) * 64 + hf * 16;
    v2i u0 = *(const v2i*)(qr +  0), u1 = *(const v2i*)(qr + 8);
    v2i u2 = *(const v2i*)(qr + 32), u3 = *(const v2i*)(qr + 40);
    v4i lo = __builtin_shufflevector(u0, u1, 0, 1, 2, 3);
    v4i hi = __builtin_shufflevector(u2, u3, 0, 1, 2, 3);
    v8i a = __builtin_shufflevector(lo, hi, 0, 1, 2, 3, 4, 5, 6, 7);
    const unsigned char* kr = sK + (size_t)(tni * 16 + r) * 64;
    v4i b0 = *(const v4i*)(kr + hf * 16);
    v4i b1 = *(const v4i*)(kr + 32 + hf * 16);
    v8i b = __builtin_shufflevector(b0, b1, 0, 1, 2, 3, 4, 5, 6, 7);
    v8i cz = {};
    v8i acc = __builtin_amdgcn_wmma_i32_16x16x64_iu8(false, a, false, b, cz, false, false);
    unsigned char* ar = sA + (size_t)(tmi * 16 + hf * 8) * 256 + tni * 16 + r;
#pragma unroll
    for (int g = 0; g < 8; ++g) ar[(size_t)g * 256] = (unsigned char)acc[g];
  }
  __syncthreads();

  float* ob = O + (size_t)tb * NN * CC + h * DD;
  for (int tt = wave; tt < 64; tt += 8) {
    const int tmi = tt >> 2, tdi = tt & 3;
    v8i acc = {};
#pragma unroll
    for (int ks = 0; ks < 4; ++ks) {
      const unsigned char* ar = sA + (size_t)(tmi * 16 + r) * 256 + ks * 64 + hf * 16;
      v2i u0 = *(const v2i*)(ar +  0), u1 = *(const v2i*)(ar + 8);
      v2i u2 = *(const v2i*)(ar + 32), u3 = *(const v2i*)(ar + 40);
      v4i lo = __builtin_shufflevector(u0, u1, 0, 1, 2, 3);
      v4i hi = __builtin_shufflevector(u2, u3, 0, 1, 2, 3);
      v8i a = __builtin_shufflevector(lo, hi, 0, 1, 2, 3, 4, 5, 6, 7);
      const unsigned char* vr = sV + (size_t)(tdi * 16 + r) * 256 + ks * 64;
      v4i b0 = *(const v4i*)(vr + hf * 16);
      v4i b1 = *(const v4i*)(vr + 32 + hf * 16);
      v8i b = __builtin_shufflevector(b0, b1, 0, 1, 2, 3, 4, 5, 6, 7);
      acc = __builtin_amdgcn_wmma_i32_16x16x64_iu8(false, a, false, b, acc, false, false);
    }
#pragma unroll
    for (int g = 0; g < 8; ++g) {
      const int n = tmi * 16 + hf * 8 + g;
      const int d = tdi * 16 + r;
      ob[(size_t)n * CC + d] = (float)acc[g] * 0.125f;
    }
  }
}

extern "C" void kernel_launch(void* const* d_in, const int* in_sizes, int n_in,
                              void* d_out, int out_size, void* d_ws, size_t ws_size,
                              hipStream_t stream) {
  (void)in_sizes; (void)n_in; (void)out_size; (void)ws_size;
  const float* x = (const float*)d_in[0];
  const float* W[6][6];
  for (int s = 0; s < 6; ++s)
    for (int j = 0; j < 6; ++j)
      W[s][j] = (const float*)d_in[1 + s * 6 + j];
  const int Kdim[6] = {CC, CC, CC, CC, CC, HD};
  const int Ndim[6] = {CC, CC, CC, CC, HD, CC};

  char* base = (char*)d_ws;
  size_t off = 0;
  auto alloc = [&](size_t bytes) -> void* {
    void* r = base + off;
    off = (off + bytes + 255) & ~(size_t)255;
    return r;
  };
  _Float16* Wt[6]; float* bias[6];
  for (int s = 0; s < 6; ++s) {
    Wt[s]  = (_Float16*)alloc((size_t)Kdim[s] * Ndim[s] * sizeof(_Float16));
    bias[s] = (float*)alloc((size_t)Ndim[s] * sizeof(float));
  }
  _Float16* xh   = (_Float16*)alloc((size_t)MROWS * CC * sizeof(_Float16));
  float*    ybuf = (float*)   alloc((size_t)MROWS * HD * sizeof(float));
  unsigned char* q8  = (unsigned char*)alloc((size_t)MROWS * CC);
  unsigned char* k8  = (unsigned char*)alloc((size_t)MROWS * CC);
  unsigned char* vT8 = (unsigned char*)alloc((size_t)MROWS * CC);
  float*    obuf = (float*)   alloc((size_t)MROWS * CC * sizeof(float));
  _Float16* sb16 = (_Float16*)alloc((size_t)MROWS * CC * sizeof(_Float16));
  float*    x1   = (float*)   alloc((size_t)MROWS * CC * sizeof(float));
  _Float16* x1h  = (_Float16*)alloc((size_t)MROWS * CC * sizeof(_Float16));
  _Float16* hbuf = (_Float16*)alloc((size_t)MROWS * HD * sizeof(_Float16));

  const dim3 blk1(256);
  const dim3 gblk(32, 4);

  for (int s = 0; s < 6; ++s) {
    size_t kn = (size_t)Kdim[s] * Ndim[s];
    fold_bn_kernel<<<dim3((unsigned)((kn + 255) / 256)), blk1, 0, stream>>>(
        W[s][0], W[s][1], W[s][2], W[s][3], W[s][4], W[s][5],
        Wt[s], bias[s], Kdim[s], Ndim[s]);
  }
  {
    size_t n = (size_t)MROWS * CC;
    cvt_f16_kernel<<<dim3((unsigned)((n + 255) / 256)), blk1, 0, stream>>>(x, xh, n);
  }
  const unsigned lifgC = (unsigned)((EE + 255) / 256);
  const unsigned lifgH = (unsigned)((EEH + 255) / 256);

  gemm_f16_kernel<<<dim3(MROWS / 128, CC / 64), gblk, 0, stream>>>(xh, Wt[0], bias[0], ybuf, CC, CC);
  lif_u8_kernel<<<lifgC, blk1, 0, stream>>>(ybuf, q8, 1.0f, EE);
  gemm_f16_kernel<<<dim3(MROWS / 128, CC / 64), gblk, 0, stream>>>(xh, Wt[1], bias[1], ybuf, CC, CC);
  lif_u8_kernel<<<lifgC, blk1, 0, stream>>>(ybuf, k8, 1.0f, EE);
  gemm_f16_kernel<<<dim3(MROWS / 128, CC / 64), gblk, 0, stream>>>(xh, Wt[2], bias[2], ybuf, CC, CC);
  lif_u8T_kernel<<<lifgC, blk1, 0, stream>>>(ybuf, vT8, 1.0f);

  attn_iu8_kernel<<<dim3(TT * BB * HH), dim3(256), 114688, stream>>>(q8, k8, vT8, obuf);
  lif_f16_kernel<<<lifgC, blk1, 0, stream>>>(obuf, sb16, 0.5f, EE);

  gemm_f16_kernel<<<dim3(MROWS / 128, CC / 64), gblk, 0, stream>>>(sb16, Wt[3], bias[3], ybuf, CC, CC);
  lif_add_kernel<<<lifgC, blk1, 0, stream>>>(ybuf, x, x1, x1h, 1.0f, EE);

  gemm_f16_kernel<<<dim3(MROWS / 128, HD / 64), gblk, 0, stream>>>(x1h, Wt[4], bias[4], ybuf, CC, HD);
  lif_f16_kernel<<<lifgH, blk1, 0, stream>>>(ybuf, hbuf, 1.0f, EEH);
  gemm_f16_kernel<<<dim3(MROWS / 128, CC / 64), gblk, 0, stream>>>(hbuf, Wt[5], bias[5], ybuf, HD, CC);
  lif_final_kernel<<<lifgC, blk1, 0, stream>>>(ybuf, x1, (float*)d_out, 1.0f, EE);
}
